// TaskEncoderSSL_19834158973625
// MI455X (gfx1250) — matrix operand order repaired and checked by reading, not compiled, not run on hardware
//
#include <hip/hip_runtime.h>
#include <hip/hip_bf16.h>

// ---------------------------------------------------------------------------
// Problem constants (from reference)
// ---------------------------------------------------------------------------
#define E_DIM   512          // E = FEAT*EMB
#define NCLS    128          // NUM_CLASS
#define MAXSHOT 1024
#define NROWS   (NCLS * MAXSHOT)   // 131072
#define EMB_D   64
#define V_DIM   32
#define TASK_D  64
#define K_CL    32

// GEMM tiling: 64 rows/block = 4 stripes of 16; 8 waves = 2 stripe-pairs x 4 col quarters
#define ROWS 64

typedef __attribute__((ext_vector_type(16))) __bf16 v16bf;
typedef __attribute__((ext_vector_type(8)))  float  v8f;

struct alignas(16) U4 { unsigned int a, b, c, d; };
union BFrag { v16bf v; U4 q[2]; };

// ---------------------------------------------------------------------------
// Workspace zeroing (cnt + ssum + ssq), every launch (determinism)
// ---------------------------------------------------------------------------
__global__ __launch_bounds__(256) void zero_f32(float* __restrict__ p, int n) {
  int i = blockIdx.x * blockDim.x + threadIdx.x;
  if (i < n) p[i] = 0.0f;
}

// ---------------------------------------------------------------------------
// W (512x512 f32, [k][n] row-major) -> bf16 transposed [n][k]
// ---------------------------------------------------------------------------
__global__ __launch_bounds__(256) void convert_w_t(const float* __restrict__ W,
                                                   __bf16* __restrict__ Wt) {
  int idx = blockIdx.x * blockDim.x + threadIdx.x;   // 512*512 threads
  int k = idx >> 9;
  int n = idx & (E_DIM - 1);
  Wt[n * E_DIM + k] = (__bf16)W[k * E_DIM + n];
}

// ---------------------------------------------------------------------------
// Fused: h1 = relu(x@W0+b0); h = h1@W1+b1; atomic segment stats on h.
// Block: 256 threads = 8 waves. Wave w: sp = w&1 selects stripe pair
// {2sp, 2sp+1} (2x16 rows), quart = w>>1 selects 128 cols (8 WMMA tiles).
// Each B fragment is loaded once and feeds two WMMAs (both stripes).
// ---------------------------------------------------------------------------
__global__ __launch_bounds__(256, 1) void fused_gemm_stats(
    const float* __restrict__ x, const int* __restrict__ y,
    const __bf16* __restrict__ W0t, const float* __restrict__ b0,
    const __bf16* __restrict__ W1t, const float* __restrict__ b1,
    float* __restrict__ cnt, float* __restrict__ ssum, float* __restrict__ ssq) {
  __shared__ __bf16 tile[ROWS * E_DIM];   // 64 KB: x tile, then reused for h1

  const int tid   = threadIdx.x;
  const int wave  = tid >> 5;
  const int lane  = tid & 31;
  const int n15   = lane & 15;
  const int hi    = lane >> 4;
  const int sp    = wave & 1;         // stripe pair: rows [32*sp, 32*sp+32)
  const int quart = wave >> 1;        // 128-column quarter
  const int row0  = (2 * sp + 0) * 16 + n15;   // A row for stripe 0 of pair
  const int row1  = (2 * sp + 1) * 16 + n15;   // A row for stripe 1 of pair
  const long rowBase = (long)blockIdx.x * ROWS;

  // ---- stage x tile (f32 -> bf16) into LDS, coalesced float4 loads ----
  for (int it = 0; it < (ROWS * E_DIM) / (256 * 4); ++it) {
    const int idx = (it * 256 + tid) * 4;
    const int r = idx >> 9;
    const int c = idx & (E_DIM - 1);
    const float4 f = *(const float4*)(x + (rowBase + r) * (long)E_DIM + c);
    __bf16* p = &tile[r * E_DIM + c];
    p[0] = (__bf16)f.x; p[1] = (__bf16)f.y; p[2] = (__bf16)f.z; p[3] = (__bf16)f.w;
  }
  __syncthreads();

  v8f zero = {};
  v8f acc0[8], acc1[8];

  // Shared GEMM body: C[pair] += tile(rows) x Wt^T, B frag reused for both stripes.
  auto run_gemm = [&](const __bf16* __restrict__ Wt) {
    for (int kt = 0; kt < E_DIM; kt += 32) {
      BFrag a0, a1;  // A 16x32 bf16: lane->row; VGPR0-3: K=hi*8+j, VGPR4-7: K=16+hi*8+j
      const __bf16* ap0 = &tile[row0 * E_DIM + kt + hi * 8];
      const __bf16* ap1 = &tile[row1 * E_DIM + kt + hi * 8];
      a0.q[0] = *(const U4*)(ap0); a0.q[1] = *(const U4*)(ap0 + 16);
      a1.q[0] = *(const U4*)(ap1); a1.q[1] = *(const U4*)(ap1 + 16);
#pragma unroll
      for (int ct = 0; ct < 8; ++ct) {
        const int n = quart * 128 + ct * 16 + n15;
        BFrag b;  // B 32x16 bf16: lane->col n; VGPR0-3: K=hi*8+j, VGPR4-7: K=16+hi*8+j
        const __bf16* bp = Wt + n * E_DIM + kt + hi * 8;
        b.q[0] = *(const U4*)(bp); b.q[1] = *(const U4*)(bp + 16);
        acc0[ct] = __builtin_amdgcn_wmma_f32_16x16x32_bf16(
            false, a0.v, false, b.v, (short)0, acc0[ct], false, false);
        acc1[ct] = __builtin_amdgcn_wmma_f32_16x16x32_bf16(
            false, a1.v, false, b.v, (short)0, acc1[ct], false, false);
      }
    }
  };

  // ================= GEMM1: relu(x @ W0 + b0) =================
#pragma unroll
  for (int ct = 0; ct < 8; ++ct) { acc0[ct] = zero; acc1[ct] = zero; }
  run_gemm(W0t);
  __syncthreads();

  // h1 = relu(acc + b0) -> bf16 back into the same LDS tile (D layout: VGPR v -> M=8*hi+v)
#pragma unroll
  for (int ct = 0; ct < 8; ++ct) {
    const int col = quart * 128 + ct * 16 + n15;
    const float bias = b0[col];
#pragma unroll
    for (int v = 0; v < 8; ++v) {
      float h0 = acc0[ct][v] + bias;
      float h1 = acc1[ct][v] + bias;
      h0 = h0 > 0.0f ? h0 : 0.0f;
      h1 = h1 > 0.0f ? h1 : 0.0f;
      tile[((2 * sp + 0) * 16 + hi * 8 + v) * E_DIM + col] = (__bf16)h0;
      tile[((2 * sp + 1) * 16 + hi * 8 + v) * E_DIM + col] = (__bf16)h1;
    }
  }
  __syncthreads();

  // ================= GEMM2: h1 @ W1 + b1 =================
#pragma unroll
  for (int ct = 0; ct < 8; ++ct) { acc0[ct] = zero; acc1[ct] = zero; }
  run_gemm(W1t);

  // ---- segment stats: f32 no-return atomics into L2-resident accumulators ----
#pragma unroll
  for (int ct = 0; ct < 8; ++ct) {
    const int col = quart * 128 + ct * 16 + n15;
    const float bias = b1[col];
#pragma unroll
    for (int v = 0; v < 8; ++v) {
      const int r0 = (2 * sp + 0) * 16 + hi * 8 + v;
      const int r1 = (2 * sp + 1) * 16 + hi * 8 + v;
      const int cls0 = y[rowBase + r0];
      const int cls1 = y[rowBase + r1];
      const float h0 = acc0[ct][v] + bias;
      const float h1 = acc1[ct][v] + bias;
      atomicAdd(&ssum[cls0 * E_DIM + col], h0);
      atomicAdd(&ssq[cls0 * E_DIM + col], h0 * h0);
      atomicAdd(&ssum[cls1 * E_DIM + col], h1);
      atomicAdd(&ssq[cls1 * E_DIM + col], h1 * h1);
      if (ct == 0 && n15 == 0 && quart == 0) {
        atomicAdd(&cnt[cls0], 1.0f);
        atomicAdd(&cnt[cls1], 1.0f);
      }
    }
  }
}

// ---------------------------------------------------------------------------
// Everything after the big GEMMs: ~10 MFLOP total, single block.
// ---------------------------------------------------------------------------
__global__ __launch_bounds__(256) void task_tail(
    const float* __restrict__ cnt, const float* __restrict__ ssum,
    const float* __restrict__ ssq,
    const float* __restrict__ Ws,  const float* __restrict__ bs,
    const float* __restrict__ Wv0, const float* __restrict__ bv0,
    const float* __restrict__ Wv1, const float* __restrict__ bv1,
    const float* __restrict__ W2,  const float* __restrict__ b2,
    const float* __restrict__ mem, float* __restrict__ out) {
  __shared__ float bufA[NCLS * EMB_D];   // 32 KB: sS, later overlaid by vS
  __shared__ float bufB[NCLS * V_DIM];   // 16 KB: v0S
  __shared__ float cntS[NCLS], invCntS[NCLS], NcS[NCLS], normS[NCLS];
  __shared__ float red[256];
  __shared__ float vmeanS[V_DIM], vvarS[V_DIM], taskS[TASK_D], rS[K_CL], scal[4];
  const int t = threadIdx.x;

  // Phase 0: per-class scalars + ||mean|| norms
  if (t < NCLS) {
    const float c = cnt[t];
    cntS[t] = c;
    invCntS[t] = 1.0f / c;
    NcS[t] = (c - 1.0f) / (float)(MAXSHOT - 1);
    float nrm = 0.0f;
    for (int j = 0; j < E_DIM; ++j) {
      const float m = ssum[t * E_DIM + j] / c;
      nrm += m * m;
    }
    normS[t] = sqrtf(nrm);
  }
  __syncthreads();

  // Phase 1: s = relu(concat(var, mean, Nc) @ Ws + bs)  -> bufA[128][64]
  for (int o = t; o < NCLS * EMB_D; o += 256) {
    const int c = o >> 6, j = o & 63;
    const float cc = cntS[c], ic = invCntS[c], icm1 = 1.0f / (cc - 1.0f);
    float a = bs[j];
    for (int d = 0; d < E_DIM; ++d) {
      const float m  = ssum[c * E_DIM + d] * ic;
      const float va = (ssq[c * E_DIM + d] - cc * m * m) * icm1;
      a += va * Ws[d * EMB_D + j] + m * Ws[(E_DIM + d) * EMB_D + j];
    }
    a += NcS[c] * Ws[(2 * E_DIM) * EMB_D + j];
    bufA[c * EMB_D + j] = a > 0.0f ? a : 0.0f;
  }
  __syncthreads();

  // Phase 2a: v0 = relu(s @ Wv0 + bv0) -> bufB[128][32]
  for (int o = t; o < NCLS * V_DIM; o += 256) {
    const int c = o >> 5, k = o & 31;
    float a = bv0[k];
    for (int j = 0; j < EMB_D; ++j) a += bufA[c * EMB_D + j] * Wv0[j * V_DIM + k];
    bufB[c * V_DIM + k] = a > 0.0f ? a : 0.0f;
  }
  __syncthreads();

  // Phase 2b: v = v0 @ Wv1 + bv1 -> overlay into bufA[0..4095]
  float* vS = bufA;
  for (int o = t; o < NCLS * V_DIM; o += 256) {
    const int c = o >> 5, k = o & 31;
    float a = bv1[k];
    for (int j = 0; j < V_DIM; ++j) a += bufB[c * V_DIM + j] * Wv1[j * V_DIM + k];
    vS[c * V_DIM + k] = a;
  }
  __syncthreads();

  // Phase 3: v stats + v_N + triu-weighted cosine sum
  if (t < V_DIM) {
    float m = 0.0f;
    for (int c = 0; c < NCLS; ++c) m += vS[c * V_DIM + t];
    m *= (1.0f / NCLS);
    float va = 0.0f;
    for (int c = 0; c < NCLS; ++c) { const float d = vS[c * V_DIM + t] - m; va += d * d; }
    vmeanS[t] = m;
    vvarS[t]  = va / (float)(NCLS - 1);
  }
  if (t == 32) {
    float a = 0.0f;
    for (int c = 0; c < NCLS; ++c) a += NcS[c];
    scal[0] = a / (float)NCLS;   // v_N
  }
  float part = 0.0f;
  for (int p = t; p < NCLS * NCLS; p += 256) {
    const int i = p >> 7, j = p & (NCLS - 1);
    if (j > i) {
      float dot = 0.0f;
      for (int d = 0; d < E_DIM; ++d) dot += ssum[i * E_DIM + d] * ssum[j * E_DIM + d];
      dot *= invCntS[i] * invCntS[j];                 // mean_i . mean_j
      const float ni = fmaxf(normS[i], 1e-7f);
      const float nj = fmaxf(normS[j], 1e-7f);
      part += (float)(j - i) * dot / (ni * nj);
    }
  }
  red[t] = part;
  __syncthreads();
  for (int st = 128; st > 0; st >>= 1) {
    if (t < st) red[t] += red[t + st];
    __syncthreads();
  }
  if (t == 0) scal[1] = red[0];   // sim_sum
  __syncthreads();

  // Phase 4: task = relu(vv @ W2 + b2), vv = [v_var(32), v_mean(32), v_N, sim_sum]
  if (t < TASK_D) {
    float a = b2[t];
    for (int d = 0; d < V_DIM; ++d) a += vvarS[d]  * W2[d * TASK_D + t];
    for (int d = 0; d < V_DIM; ++d) a += vmeanS[d] * W2[(V_DIM + d) * TASK_D + t];
    a += scal[0] * W2[(2 * V_DIM)     * TASK_D + t];
    a += scal[1] * W2[(2 * V_DIM + 1) * TASK_D + t];
    taskS[t] = a > 0.0f ? a : 0.0f;
  }
  __syncthreads();

  // Phase 5: Student-t responsibilities over mem clusters
  if (t < K_CL) {
    float dd = 0.0f;
    for (int j = 0; j < TASK_D; ++j) {
      const float d = taskS[j] - mem[t * TASK_D + j];
      dd += d * d;
    }
    const float dist = sqrtf(dd);
    rS[t] = powf(dist / 1.0f + 1.0f, -(1.0f + 1.0f) * 0.5f);   // TEMP = 1
  }
  __syncthreads();
  if (t == 0) {
    float a = 0.0f;
    for (int k = 0; k < K_CL; ++k) a += rS[k];
    scal[2] = a;
  }
  __syncthreads();

  // Phase 6: out = C @ mem
  if (t < TASK_D) {
    const float inv = 1.0f / scal[2];
    float a = 0.0f;
    for (int k = 0; k < K_CL; ++k) a += rS[k] * inv * mem[k * TASK_D + t];
    out[t] = a;
  }
}

// ---------------------------------------------------------------------------
// Launch
// ---------------------------------------------------------------------------
extern "C" void kernel_launch(void* const* d_in, const int* in_sizes, int n_in,
                              void* d_out, int out_size, void* d_ws, size_t ws_size,
                              hipStream_t stream) {
  (void)in_sizes; (void)n_in; (void)out_size; (void)ws_size;
  const float* x   = (const float*)d_in[0];
  const int*   y   = (const int*)  d_in[1];
  const float* W0  = (const float*)d_in[2];
  const float* b0  = (const float*)d_in[3];
  const float* W1  = (const float*)d_in[4];
  const float* b1  = (const float*)d_in[5];
  const float* Ws  = (const float*)d_in[6];
  const float* bs  = (const float*)d_in[7];
  const float* Wv0 = (const float*)d_in[8];
  const float* bv0 = (const float*)d_in[9];
  const float* Wv1 = (const float*)d_in[10];
  const float* bv1 = (const float*)d_in[11];
  const float* W2  = (const float*)d_in[12];
  const float* b2  = (const float*)d_in[13];
  const float* mem = (const float*)d_in[14];
  float* out = (float*)d_out;

  // workspace layout: W0t (512KB) | W1t (512KB) | cnt (512B) | ssum (256KB) | ssq (256KB)
  char* ws = (char*)d_ws;
  __bf16* W0t = (__bf16*)ws;
  __bf16* W1t = (__bf16*)(ws + (size_t)E_DIM * E_DIM * 2);
  float*  cnt = (float*)(ws + (size_t)2 * E_DIM * E_DIM * 2);
  float*  ssum = cnt + NCLS;
  float*  ssq  = ssum + (size_t)NCLS * E_DIM;

  const int statN = NCLS + 2 * NCLS * E_DIM;
  zero_f32<<<(statN + 255) / 256, 256, 0, stream>>>(cnt, statN);
  convert_w_t<<<(E_DIM * E_DIM) / 256, 256, 0, stream>>>(W0, W0t);
  convert_w_t<<<(E_DIM * E_DIM) / 256, 256, 0, stream>>>(W1, W1t);
  fused_gemm_stats<<<NROWS / ROWS, 256, 0, stream>>>(x, y, W0t, b0, W1t, b1,
                                                     cnt, ssum, ssq);
  task_tail<<<1, 256, 0, stream>>>(cnt, ssum, ssq, Ws, bs, Wv0, bv0, Wv1, bv1,
                                   W2, b2, mem, out);
}
